// STBlock_86775519249025
// MI455X (gfx1250) — matrix operand order repaired and checked by reading, not compiled, not run on hardware
//
#include <hip/hip_runtime.h>
#include <hip/hip_bf16.h>

// ---------------------------------------------------------------------------
// Types
// ---------------------------------------------------------------------------
typedef __attribute__((ext_vector_type(16))) __bf16 v16bf;
typedef __attribute__((ext_vector_type(8)))  float  v8f;
typedef __attribute__((ext_vector_type(4)))  unsigned int u32x4;

union Frag { v16bf v; u32x4 q[2]; };

__device__ __forceinline__ unsigned short f32_to_bf16(float f) {
    unsigned int u = __float_as_uint(f);
    unsigned int r = u + 0x7fffu + ((u >> 16) & 1u);   // round-to-nearest-even
    return (unsigned short)(r >> 16);
}

// ---------------------------------------------------------------------------
// JAX threefry2x32 dropout mask: key(42) = {0,42}, p_keep = 0.7
// mask over flat row-major index e of x (S = 8*64*1024*12 = 6291456)
// ---------------------------------------------------------------------------
__device__ __forceinline__ unsigned int rotl32(unsigned int v, int r) {
    return (v << r) | (v >> (32 - r));
}

__device__ float dropout_scale(unsigned int e) {
    const unsigned int S_half = 3145728u;          // S / 2
    const bool hi = (e >= S_half);
    unsigned int i = hi ? (e - S_half) : e;
    unsigned int x0 = i;
    unsigned int x1 = S_half + i;
    const unsigned int k0 = 0u, k1 = 42u;
    const unsigned int k2 = k0 ^ k1 ^ 0x1BD11BDAu;
    x0 += k0; x1 += k1;
#define TF_R(r) { x0 += x1; x1 = rotl32(x1, r); x1 ^= x0; }
#define TF_G0()  TF_R(13) TF_R(15) TF_R(26) TF_R(6)
#define TF_G1()  TF_R(17) TF_R(29) TF_R(16) TF_R(24)
    TF_G0(); x0 += k1; x1 += k2 + 1u;
    TF_G1(); x0 += k2; x1 += k0 + 2u;
    TF_G0(); x0 += k0; x1 += k1 + 3u;
    TF_G1(); x0 += k1; x1 += k2 + 4u;
    TF_G0(); x0 += k2; x1 += k0 + 5u;
#undef TF_R
#undef TF_G0
#undef TF_G1
    unsigned int bits = hi ? x1 : x0;
    float u = __uint_as_float((bits >> 9) | 0x3f800000u) - 1.0f;  // [0,1)
    return (u < 0.7f) ? (1.0f / 0.7f) : 0.0f;
}

// ---------------------------------------------------------------------------
// prep_adj: M = 0.75*I + 0.25 * D^-1 (A + I), emitted BOTH row-major (M) and
// transposed (MT) in bf16 so every later GEMM stages straight row-major tiles.
// transpose=1 builds the backward direction (from adj^T). Block per row v.
// ---------------------------------------------------------------------------
__global__ __launch_bounds__(256) void prep_adj(const float* __restrict__ adj,
                                                unsigned short* __restrict__ M,
                                                unsigned short* __restrict__ MT,
                                                int transpose) {
    __shared__ float red[256];
    const int v = blockIdx.x;
    const int tid = threadIdx.x;
    float s = 0.f;
    for (int w = tid; w < 1024; w += 256)
        s += transpose ? adj[(size_t)w * 1024 + v] : adj[(size_t)v * 1024 + w];
    red[tid] = s;
    __syncthreads();
    for (int off = 128; off > 0; off >>= 1) {
        if (tid < off) red[tid] += red[tid + off];
        __syncthreads();
    }
    const float inv = 0.25f / (red[0] + 1.0f);   // rowsum includes self-loop
    for (int w = tid; w < 1024; w += 256) {
        float a = transpose ? adj[(size_t)w * 1024 + v] : adj[(size_t)v * 1024 + w];
        float val = a * inv + ((w == v) ? (0.75f + inv) : 0.f);
        unsigned short h = f32_to_bf16(val);
        M [(size_t)v * 1024 + w] = h;
        MT[(size_t)w * 1024 + v] = h;
    }
}

// ---------------------------------------------------------------------------
// chanmix_dropout: G2[j=(b,o,t)][w] = sum_c W[o][c] * dropout(x)[b][c][w][t]
// (channel mix commutes with graph diffusion). Stored row-major [j][w] so it
// is the A-operand of the flipped final GEMM; stores are w-contiguous.
// Grid: 8(b) * 12(t) * 8(w-tiles of 128) = 768 blocks, 256 threads.
// ---------------------------------------------------------------------------
__global__ __launch_bounds__(256) void chanmix_dropout(
        const float* __restrict__ x, const float* __restrict__ W,
        unsigned short* __restrict__ G2) {
    __shared__ float sW[64 * 64];
    __shared__ float sX[64 * 128];
    const int blk = blockIdx.x;
    const int b  = blk / 96;
    const int r  = blk - b * 96;
    const int t  = r >> 3;
    const int w0 = (r & 7) << 7;
    const int tid = threadIdx.x;
    for (int i = tid; i < 4096; i += 256) sW[i] = W[i];
    for (int i = tid; i < 8192; i += 256) {
        int c = i >> 7, w = i & 127;
        unsigned int e = ((unsigned)(b * 64 + c) * 1024u + (unsigned)(w0 + w)) * 12u
                         + (unsigned)t;
        sX[i] = x[e] * dropout_scale(e);
    }
    __syncthreads();
    for (int i = tid; i < 8192; i += 256) {
        int o = i >> 7, w = i & 127;
        float acc = 0.f;
#pragma unroll 16
        for (int c = 0; c < 64; ++c)
            acc = fmaf(sW[o * 64 + c], sX[c * 128 + w], acc);
        int j = b * 768 + o * 12 + t;
        G2[(size_t)j * 1024 + w0 + w] = f32_to_bf16(acc);
    }
}

// ---------------------------------------------------------------------------
// Tiled bf16 WMMA GEMM with async double-buffered LDS staging.
//   C[m][n] = sum_k A[m][k] * BT[n][k]        (B given pre-transposed)
// Block tile 128(M) x 256(N) x 64(K); 256 threads = 8 waves in a 2x4 grid,
// each wave owns a 64x64 patch = 4x4 v_wmma_f32_16x16x32_bf16 accumulators
// (16 WMMA per k-step vs 16 ds_load_b128 fragment loads, 4x operand reuse).
// Both tiles are row-major copies staged with GLOBAL_LOAD_ASYNC_TO_LDS_B128
// (ASYNCcnt), double-buffered with a single barrier per K iteration.
// mode 0: C -> bf16 row-major AND CT -> bf16 transposed (matrix squaring)
// mode 1: out[b,o,v,t]  = acc + bias[o]   (rows m are j=(b,o,t), cols are v)
// mode 2: out[b,o,v,t] += acc + bias[o]
// ---------------------------------------------------------------------------
#define BM 128
#define BN 256
#define BK 64
#define LSTR 72   // ushorts per LDS row: 64 data + 8 pad (144 B, conflict-free b128)

__device__ __forceinline__ void async_b128(unsigned lds_byte_off,
                                           unsigned long long gbase,
                                           unsigned gbyte_off) {
    // GVS mode: mem = SADDR(64) + VADDR(32) ; VDST = LDS byte address
    asm volatile("global_load_async_to_lds_b128 %0, %1, %2"
                 :: "v"(lds_byte_off), "v"(gbyte_off), "s"(gbase)
                 : "memory");
}

__global__ __launch_bounds__(256) void gemm_wmma(
        const unsigned short* __restrict__ A,    // [M][K] bf16 row-major
        const unsigned short* __restrict__ BT,   // [N][K] bf16 row-major
        unsigned short* __restrict__ C,          // mode 0
        unsigned short* __restrict__ CT,         // mode 0
        float* __restrict__ outF,                // mode 1/2
        const float* __restrict__ bias,
        int K, int ldc, int mode) {
    __shared__ unsigned short sA[2][BM * LSTR];
    __shared__ unsigned short sB[2][BN * LSTR];

    const int tid  = threadIdx.x;
    const int lane = tid & 31;
    const int wid  = tid >> 5;
    const int wm   = wid >> 2;      // 0..1  (64 rows each)
    const int wn   = wid & 3;       // 0..3  (64 cols each)
    const int m0   = blockIdx.y * BM;
    const int n0   = blockIdx.x * BN;
    const int l15  = lane & 15;
    const int hi   = lane >> 4;

    const unsigned ldsA0 = (unsigned)(unsigned long long)(&sA[0][0]);
    const unsigned ldsB0 = (unsigned)(unsigned long long)(&sB[0][0]);
    const unsigned bufA = BM * LSTR * 2u;
    const unsigned bufB = BN * LSTR * 2u;
    const unsigned long long aBase = (unsigned long long)A;
    const unsigned long long bBase = (unsigned long long)BT;

    // Per stage: A tile 128x64 (4 b128/thread), B tile 256x64 (8 b128/thread).
    auto issue_stage = [&](int kb, int buf) {
#pragma unroll
        for (int p = 0; p < 4; ++p) {
            int idx = p * 256 + tid;
            int row = idx >> 3, ch = idx & 7;
            async_b128(ldsA0 + buf * bufA + (unsigned)(row * LSTR + ch * 8) * 2u,
                       aBase, (unsigned)((m0 + row) * K + kb + ch * 8) * 2u);
        }
#pragma unroll
        for (int p = 0; p < 8; ++p) {
            int idx = p * 256 + tid;
            int row = idx >> 3, ch = idx & 7;
            async_b128(ldsB0 + buf * bufB + (unsigned)(row * LSTR + ch * 8) * 2u,
                       bBase, (unsigned)((n0 + row) * K + kb + ch * 8) * 2u);
        }
    };

    v8f acc[4][4];
    const v8f vzero = {};
#pragma unroll
    for (int i = 0; i < 4; ++i)
#pragma unroll
        for (int j = 0; j < 4; ++j) acc[i][j] = vzero;

    issue_stage(0, 0);
    const int nkb = K / BK;
    for (int ib = 0; ib < nkb; ++ib) {
        const int buf = ib & 1;
        asm volatile("s_wait_asynccnt 0x0" ::: "memory");  // this wave's DMAs done
        __syncthreads();   // all DMAs done AND buf^1 no longer being read
        if (ib + 1 < nkb) issue_stage((ib + 1) * BK, buf ^ 1);

#pragma unroll
        for (int ks = 0; ks < BK; ks += 32) {
            Frag af[4], bf[4];
            // A frag (16-bit 16x32 layout): lane<16 -> K 0-7 & 16-23; lane>=16 -> 8-15 & 24-31
#pragma unroll
            for (int i = 0; i < 4; ++i) {
                int row = wm * 64 + i * 16 + l15;
                const unsigned short* base = &sA[buf][row * LSTR + ks + hi * 8];
                af[i].q[0] = *(const u32x4*)(base);
                af[i].q[1] = *(const u32x4*)(base + 16);
            }
            // B frag: lane<16 -> col n, K 0-7 & 16-23; lane>=16 -> col n, K 8-15 & 24-31
#pragma unroll
            for (int j = 0; j < 4; ++j) {
                int col = wn * 64 + j * 16 + l15;
                const unsigned short* base = &sB[buf][col * LSTR + ks + hi * 8];
                bf[j].q[0] = *(const u32x4*)(base);
                bf[j].q[1] = *(const u32x4*)(base + 16);
            }
#pragma unroll
            for (int i = 0; i < 4; ++i)
#pragma unroll
                for (int j = 0; j < 4; ++j)
                    acc[i][j] = __builtin_amdgcn_wmma_f32_16x16x32_bf16(
                        false, af[i].v, false, bf[j].v,
                        (short)0, acc[i][j], false, false);
        }
    }

    // ---- epilogue (C/D layout: VGPR r = row +r (+8 for hi half), lane -> col) ----
    if (mode == 0) {
#pragma unroll
        for (int i = 0; i < 4; ++i)
#pragma unroll
            for (int j = 0; j < 4; ++j) {
                int coln = n0 + wn * 64 + j * 16 + l15;
#pragma unroll
                for (int r = 0; r < 8; ++r) {
                    int row = m0 + wm * 64 + i * 16 + hi * 8 + r;
                    unsigned short h = f32_to_bf16(acc[i][j][r]);
                    C [(size_t)row * ldc + coln] = h;
                    CT[(size_t)coln * ldc + row] = h;
                }
            }
    } else {
#pragma unroll
        for (int i = 0; i < 4; ++i)
#pragma unroll
            for (int j = 0; j < 4; ++j) {
                int v = n0 + wn * 64 + j * 16 + l15;        // graph node
#pragma unroll
                for (int r = 0; r < 8; ++r) {
                    int jr  = m0 + wm * 64 + i * 16 + hi * 8 + r;  // j = b*768+o*12+t
                    int bb  = jr / 768;
                    int rem = jr - bb * 768;
                    int o   = rem / 12;
                    int t   = rem - o * 12;
                    size_t oi = (size_t)bb * 786432 + (size_t)o * 12288
                              + (size_t)v * 12 + (size_t)t;
                    float val = acc[i][j][r] + bias[o];
                    if (mode == 2) val += outF[oi];
                    outF[oi] = val;
                }
            }
    }
}

// ---------------------------------------------------------------------------
// Orchestration.  out = M_fwd^4 (W1 xd) + b1  +  M_bwd^4 (W2 xd) + b2
// (Euler steps folded into M = 0.75I + 0.25*A_hat, then M^4 by two squarings;
//  channel mix commutes with diffusion so it runs first.)
// Workspace (22 MB): M | MT | M2 | M2T | M4T (2MB each) | G2 bf16 6144x1024
// ---------------------------------------------------------------------------
extern "C" void kernel_launch(void* const* d_in, const int* in_sizes, int n_in,
                              void* d_out, int out_size, void* d_ws, size_t ws_size,
                              hipStream_t stream) {
    (void)in_sizes; (void)n_in; (void)out_size; (void)ws_size;
    const float* x   = (const float*)d_in[0];
    const float* adj = (const float*)d_in[1];
    const float* W1  = (const float*)d_in[2];
    const float* b1  = (const float*)d_in[3];
    const float* W2  = (const float*)d_in[4];
    const float* b2  = (const float*)d_in[5];
    float* out = (float*)d_out;

    char* ws = (char*)d_ws;
    unsigned short* Mt  = (unsigned short*)(ws);
    unsigned short* MtT = (unsigned short*)(ws + (2ull << 20));
    unsigned short* M2  = (unsigned short*)(ws + (4ull << 20));
    unsigned short* M2T = (unsigned short*)(ws + (6ull << 20));
    unsigned short* M4T = (unsigned short*)(ws + (8ull << 20));
    unsigned short* G2  = (unsigned short*)(ws + (10ull << 20));

    for (int dir = 0; dir < 2; ++dir) {
        const float* W    = dir ? W2 : W1;
        const float* bias = dir ? b2 : b1;

        prep_adj<<<1024, 256, 0, stream>>>(adj, Mt, MtT, dir);

        // M^2 = M*M (store both orientations); M^4 = M^2*M^2 (need only M4T;
        // row-major copy lands in Mt which is dead by then).
        gemm_wmma<<<dim3(4, 8), 256, 0, stream>>>(Mt, MtT, M2, M2T, nullptr,
                                                  nullptr, 1024, 1024, 0);
        gemm_wmma<<<dim3(4, 8), 256, 0, stream>>>(M2, M2T, Mt, M4T, nullptr,
                                                  nullptr, 1024, 1024, 0);

        // G2 = W * dropout(x), bf16, rows j=(b,o,t), cols w
        chanmix_dropout<<<768, 256, 0, stream>>>(x, W, G2);

        // out (+)= (G2 * M4^T)^T + bias : M=6144 rows j, N=1024 nodes, K=1024
        gemm_wmma<<<dim3(4, 48), 256, 0, stream>>>(G2, M4T, nullptr, nullptr,
                                                   out, bias, 1024, 1024,
                                                   dir ? 2 : 1);
    }
}
